// DecoderLayer_45174466020042
// MI455X (gfx1250) — matrix operand order repaired and checked by reading, not compiled, not run on hardware
//
#include <hip/hip_runtime.h>
#include <hip/hip_bf16.h>

#define H 4096
#define NH 32
#define HD 128
#define S_LEN 2048
#define B_SZ 2
#define H3 (3*H)
#define FF (4*H)
#define EPS_LN 1e-5f
#define ATTN_SCALE 0.08838834764831845f   // HD^-0.5

typedef __attribute__((ext_vector_type(16))) __bf16 v16bf;
typedef __attribute__((ext_vector_type(8)))  __bf16 v8bf;
typedef __attribute__((ext_vector_type(8)))  float  v8f;
typedef __attribute__((ext_vector_type(4)))  unsigned int u32x4;
typedef __attribute__((ext_vector_type(8)))  int i32x8;
typedef __attribute__((ext_vector_type(4)))  int i32x4;

__device__ __forceinline__ unsigned short f2bf(float x) {
  return __builtin_bit_cast(unsigned short, (__bf16)x);
}
__device__ __forceinline__ float bf2f(unsigned short u) {
  return (float)__builtin_bit_cast(__bf16, u);
}
__device__ __forceinline__ v8bf ld8(const unsigned short* p) {
  return *reinterpret_cast<const v8bf*>(p);
}
__device__ __forceinline__ v16bf cat16(v8bf lo, v8bf hi) {
  return __builtin_shufflevector(lo, hi, 0,1,2,3,4,5,6,7,8,9,10,11,12,13,14,15);
}
__device__ __forceinline__ v8f wmma_bf16(v16bf a, v16bf b, v8f c) {
  return __builtin_amdgcn_wmma_f32_16x16x32_bf16(false, a, false, b, (short)0, c, false, false);
}

// LDS byte offset of a generic pointer to __shared__: flat LDS addresses carry
// the LDS offset in bits [31:0] (CDNA5 aperture mapping).
__device__ __forceinline__ unsigned lds_off(const void* p) {
  return (unsigned)(uintptr_t)p;
}

// Async global->LDS 16B copy (per-lane), tracked by ASYNCcnt.
__device__ __forceinline__ void async_b128(unsigned loff, const void* g) {
  asm volatile("global_load_async_to_lds_b128 %0, %1, off"
               :: "v"(loff), "v"((unsigned long long)(uintptr_t)g)
               : "memory");
}
__device__ __forceinline__ void wait_async0() {
  asm volatile("s_wait_asynccnt 0x0" ::: "memory");
}

// Tensor Data Mover: 2D tile of 16-bit elements, rows*cols, global row stride
// `gstride` (elements), into LDS at `loff` with 4-DWORD pad every 32 DWORDs
// (i.e. 64-elem rows land with LDS stride 72 elems).
__device__ __forceinline__ void tdm_load_2d_bf16(unsigned loff, const void* g,
                                                 unsigned tdim0, unsigned tdim1,
                                                 unsigned tile0, unsigned tile1,
                                                 unsigned gstride,
                                                 unsigned pad_int, unsigned pad_amt) {
  unsigned long long ga = (unsigned long long)(uintptr_t)g;
  u32x4 g0;
  g0[0] = 1u;                                    // count=1, user descriptor
  g0[1] = loff;                                  // lds_addr
  g0[2] = (unsigned)(ga & 0xFFFFFFFFu);          // global_addr[31:0]
  g0[3] = (unsigned)((ga >> 32) & 0x01FFFFFFu) | 0x80000000u; // addr[56:32] | type=2
  i32x8 g1;
  g1[0] = (int)((1u << 16) | (1u << 20) | (pad_int << 22) | (pad_amt << 25));
  g1[1] = (int)((tdim0 & 0xFFFFu) << 16);        // atomic_barrier_addr=0 | tensor_dim0 lo
  g1[2] = (int)(((tdim0 >> 16) & 0xFFFFu) | ((tdim1 & 0xFFFFu) << 16));
  g1[3] = (int)(((tdim1 >> 16) & 0xFFFFu) | (tile0 << 16));   // tile_dim0
  g1[4] = (int)tile1;                            // tile_dim1 | tile_dim2=0
  g1[5] = (int)gstride;                          // tensor_dim0_stride[31:0]
  g1[6] = 0;                                     // stride hi | dim1_stride lo
  g1[7] = 0;
  i32x4 gz = {0, 0, 0, 0};
#if defined(__clang_major__) && __clang_major__ >= 23
  i32x8 gz8 = {0, 0, 0, 0, 0, 0, 0, 0};
  __builtin_amdgcn_tensor_load_to_lds(g0, g1, gz, gz, gz8, 0);
#else
  __builtin_amdgcn_tensor_load_to_lds(g0, g1, gz, gz, 0);
#endif
}

// ---------------------------------------------------------------- LayerNorm
__global__ __launch_bounds__(256) void ln_kernel(
    const float* __restrict__ x, const float* __restrict__ w,
    const float* __restrict__ b, unsigned short* __restrict__ out) {
  __shared__ float rs[256], rq[256];
  const int row = blockIdx.x, tid = threadIdx.x;
  const float* xr = x + (size_t)row * H;
  float v[16]; float s = 0.f, q = 0.f;
  #pragma unroll
  for (int i = 0; i < 16; ++i) { v[i] = xr[i*256 + tid]; s += v[i]; q += v[i]*v[i]; }
  rs[tid] = s; rq[tid] = q; __syncthreads();
  for (int st = 128; st > 0; st >>= 1) {
    if (tid < st) { rs[tid] += rs[tid+st]; rq[tid] += rq[tid+st]; }
    __syncthreads();
  }
  const float mu = rs[0] * (1.f/H);
  const float var = rq[0] * (1.f/H) - mu*mu;
  const float rstd = rsqrtf(var + EPS_LN);
  unsigned short* orow = out + (size_t)row * H;
  #pragma unroll
  for (int i = 0; i < 16; ++i) {
    int c = i*256 + tid;
    orow[c] = f2bf((v[i]-mu)*rstd*w[c] + b[c]);
  }
}

// ---------------------------------------------------------------- GEMM (bf16 WMMA)
// Tile: 128x128x64. A (bf16) staged by TDM (double-buffered, padded layout by DMA);
// W (f32) staged via register pipeline + convert; WMMA on 16x16x32 bf16 fragments.
#define BM 128
#define BN 128
#define BK 64
#define ASTR 72   // 144B row stride, 16B aligned
#define BSTR 72

// EPI: 0 = bias -> bf16 ; 1 = gelu(bias) -> bf16 ; 2 = bias -> f32 ;
//      3 = bias + add1 + add2 -> f32
template<int EPI>
__global__ __launch_bounds__(256) void gemm_bf16_kernel(
    const unsigned short* __restrict__ A, const float* __restrict__ W,
    const float* __restrict__ bias, float* __restrict__ outF,
    unsigned short* __restrict__ outB, const float* __restrict__ add1,
    const float* __restrict__ add2, int M, int N, int K) {
  __shared__ unsigned short la[2][BM*ASTR];   // A tiles (TDM destination)
  __shared__ unsigned short lb[BN*BSTR];      // W tile, transposed [n][k]
  const int tid = threadIdx.x;
  const int lane = tid & 31, hi = lane >> 4, l15 = lane & 15;
  const int wid = tid >> 5;
  const int wm = wid & 3, wn = wid >> 2;
  const int m0 = blockIdx.y * BM, n0 = blockIdx.x * BN;
  v8f acc[2][4] = {};
  float4 wreg[8];

  auto loadW = [&](int k0) {
    #pragma unroll
    for (int c = 0; c < 8; ++c) {
      int idx = c*256 + tid;
      int kk = idx >> 5, nc = (idx & 31) * 4;
      wreg[c] = *reinterpret_cast<const float4*>(&W[(size_t)(k0 + kk)*N + n0 + nc]);
    }
  };
  auto tdmA = [&](int k0, int buf) {
    if (tid == 0) {
      // 64 elems/row * 2B = 32 DWORDs -> pad_interval code 4; pad 4 DWORDs -> code 3
      tdm_load_2d_bf16(lds_off(&la[buf][0]), &A[(size_t)m0*K + k0],
                       (unsigned)K, (unsigned)M, BK, BM, (unsigned)K, 4u, 3u);
    }
  };

  loadW(0);
  tdmA(0, 0);
  int buf = 0;
  for (int k0 = 0; k0 < K; k0 += BK) {
    // Commit W regs -> LDS (convert f32->bf16, transposed)
    #pragma unroll
    for (int c = 0; c < 8; ++c) {
      int idx = c*256 + tid;
      int kk = idx >> 5, nc = (idx & 31) * 4;
      lb[(nc+0)*BSTR + kk] = f2bf(wreg[c].x);
      lb[(nc+1)*BSTR + kk] = f2bf(wreg[c].y);
      lb[(nc+2)*BSTR + kk] = f2bf(wreg[c].z);
      lb[(nc+3)*BSTR + kk] = f2bf(wreg[c].w);
    }
    __builtin_amdgcn_s_wait_tensorcnt((short)0);   // A tile DMA complete (wave 0)
    __syncthreads();
    if (k0 + BK < K) {            // overlap next tile's traffic with compute
      loadW(k0 + BK);
      tdmA(k0 + BK, buf ^ 1);
    }
    const unsigned short* lab = la[buf];
    #pragma unroll
    for (int ks = 0; ks < BK; ks += 32) {
      v16bf af[2], bfr[4];
      #pragma unroll
      for (int i = 0; i < 2; ++i) {
        int base = (wm*32 + i*16 + l15)*ASTR + ks + hi*8;
        af[i] = cat16(ld8(&lab[base]), ld8(&lab[base + 16]));
      }
      #pragma unroll
      for (int j = 0; j < 4; ++j) {
        int base = (wn*64 + j*16 + l15)*BSTR + ks + hi*8;
        bfr[j] = cat16(ld8(&lb[base]), ld8(&lb[base + 16]));
      }
      #pragma unroll
      for (int i = 0; i < 2; ++i)
        #pragma unroll
        for (int j = 0; j < 4; ++j)
          acc[i][j] = wmma_bf16(af[i], bfr[j], acc[i][j]);
    }
    __syncthreads();
    buf ^= 1;
  }
  // Epilogue. C layout: VGPR r -> row M = r + 8*hi, col N = l15.
  #pragma unroll
  for (int i = 0; i < 2; ++i) {
    #pragma unroll
    for (int j = 0; j < 4; ++j) {
      const int col = n0 + wn*64 + j*16 + l15;
      const float bv = bias[col];
      #pragma unroll
      for (int r = 0; r < 8; ++r) {
        const int row = m0 + wm*32 + i*16 + r + 8*hi;
        const size_t oi = (size_t)row*N + col;
        float v = acc[i][j][r] + bv;
        if constexpr (EPI == 0) {
          outB[oi] = f2bf(v);
        } else if constexpr (EPI == 1) {
          outB[oi] = f2bf(0.5f * v * (1.f + erff(v * 0.70710678118654752f)));
        } else if constexpr (EPI == 2) {
          outF[oi] = v;
        } else {
          outF[oi] = v + add1[oi] + add2[oi];
        }
      }
    }
  }
}

// ---------------------------------------------------------------- RoPE
__global__ __launch_bounds__(256) void rope_kernel(
    const unsigned short* __restrict__ qkv, const long long* __restrict__ pid,
    unsigned short* __restrict__ qr, unsigned short* __restrict__ kr) {
  size_t idx = (size_t)blockIdx.x * 256 + threadIdx.x;   // over (B*S)*NH*HD
  const int d = (int)(idx & (HD-1));
  const size_t rh = idx >> 7;
  const int h = (int)(rh & (NH-1));
  const size_t row = rh >> 5;
  const float pos = (float)pid[row];
  const unsigned short* qs = qkv + row*(size_t)H3 + h*HD;
  const unsigned short* ks = qs + H;
  float oq, ok;
  if (d < 64) {
    const int i = (d < 32) ? d : d - 32;
    const float inv = __powf(10000.f, -(float)i * (1.f/32.f));
    const float fr = pos * inv;
    const float c = __cosf(fr), s = __sinf(fr);
    const float q1 = bf2f(qs[i]), q2 = bf2f(qs[i+32]);
    const float k1 = bf2f(ks[i]), k2 = bf2f(ks[i+32]);
    if (d < 32) { oq = q1*c - q2*s; ok = k1*c - k2*s; }
    else        { oq = q2*c + q1*s; ok = k2*c + k1*s; }
  } else {
    oq = bf2f(qs[d]); ok = bf2f(ks[d]);
  }
  const size_t o = row*(size_t)H + h*HD + d;
  qr[o] = f2bf(oq); kr[o] = f2bf(ok);
}

// ---------------------------------------------------------------- Flash attention
// Block = 128 threads (4 waves), one (b, h, 64-query tile) per block.
#define KSTR 136   // [j(32)][d(128)] K tile, padded
#define VSTR 40    // [d(128)][j(32)] transposed V tile, padded
#define PSTR 40    // per-wave P scratch [m(16)][j(32)]

__global__ __launch_bounds__(128) void attn_kernel(
    const unsigned short* __restrict__ q, const unsigned short* __restrict__ kmat,
    const unsigned short* __restrict__ qkv, unsigned short* __restrict__ out) {
  __shared__ unsigned short lk[32*KSTR];
  __shared__ unsigned short lv[128*VSTR];
  __shared__ unsigned short pl[4][16*PSTR];
  const int tid = threadIdx.x;
  const int lane = tid & 31, hi = lane >> 4, l15 = lane & 15;
  const int wid = tid >> 5;
  const int qb0 = blockIdx.x * 64;
  const int h = blockIdx.y;
  const int b = blockIdx.z;
  const size_t bs0 = (size_t)b * S_LEN;
  const int qrow0 = qb0 + wid*16;

  v16bf qf[4];
  {
    const unsigned short* qbase = q + (bs0 + qrow0 + l15)*(size_t)H + h*HD;
    #pragma unroll
    for (int s = 0; s < 4; ++s)
      qf[s] = cat16(ld8(qbase + s*32 + hi*8), ld8(qbase + s*32 + 16 + hi*8));
  }
  v8f acc[8] = {};
  float mrow[8], lrow[8];
  #pragma unroll
  for (int r = 0; r < 8; ++r) { mrow[r] = -1e30f; lrow[r] = 0.f; }
  unsigned short* plw = pl[wid];
  const int kend = qb0 + 64;
  for (int kb = 0; kb < kend; kb += 32) {
    // K tile: async DMA to LDS in aligned 16B chunks (ASYNCcnt)
    #pragma unroll
    for (int c = 0; c < 4; ++c) {
      int cc = c*128 + tid;
      int j = cc >> 4, d8 = (cc & 15) * 8;
      async_b128(lds_off(&lk[j*KSTR + d8]),
                 kmat + (bs0 + kb + j)*(size_t)H + h*HD + d8);
    }
    // V tile: manual transposed stage
    for (int idx = tid; idx < 32*128; idx += 128) {
      int j = idx >> 7, d = idx & 127;
      lv[d*VSTR + j] = qkv[(bs0 + kb + j)*(size_t)H3 + 2*H + h*HD + d];
    }
    wait_async0();
    __syncthreads();
    // Preload all 8 K fragments, then 8 WMMAs (one dscnt wait for the batch)
    v16bf kf[8];
    #pragma unroll
    for (int ss = 0; ss < 4; ++ss) {
      int d0 = ss*32 + hi*8;
      kf[ss*2]   = cat16(ld8(&lk[l15*KSTR + d0]),      ld8(&lk[l15*KSTR + d0 + 16]));
      kf[ss*2+1] = cat16(ld8(&lk[(16+l15)*KSTR + d0]), ld8(&lk[(16+l15)*KSTR + d0 + 16]));
    }
    v8f s0 = {}, s1 = {};
    #pragma unroll
    for (int ss = 0; ss < 4; ++ss) {
      s0 = wmma_bf16(qf[ss], kf[ss*2],   s0);
      s1 = wmma_bf16(qf[ss], kf[ss*2+1], s1);
    }
    // causal mask + online softmax (rows live in 16-lane halves)
    #pragma unroll
    for (int r = 0; r < 8; ++r) {
      const int qg = qrow0 + r + 8*hi;
      const int kg0 = kb + l15, kg1 = kg0 + 16;
      float e0 = (kg0 <= qg) ? s0[r]*ATTN_SCALE : -1e30f;
      float e1 = (kg1 <= qg) ? s1[r]*ATTN_SCALE : -1e30f;
      float t = fmaxf(e0, e1);
      t = fmaxf(t, __shfl_xor(t, 1, 32));
      t = fmaxf(t, __shfl_xor(t, 2, 32));
      t = fmaxf(t, __shfl_xor(t, 4, 32));
      t = fmaxf(t, __shfl_xor(t, 8, 32));
      const float mn = fmaxf(mrow[r], t);
      const float cf = __expf(mrow[r] - mn);
      mrow[r] = mn;
      const float p0 = __expf(e0 - mn), p1 = __expf(e1 - mn);
      float rsum = p0 + p1;
      rsum += __shfl_xor(rsum, 1, 32);
      rsum += __shfl_xor(rsum, 2, 32);
      rsum += __shfl_xor(rsum, 4, 32);
      rsum += __shfl_xor(rsum, 8, 32);
      lrow[r] = lrow[r]*cf + rsum;
      #pragma unroll
      for (int t8 = 0; t8 < 8; ++t8) acc[t8][r] *= cf;
      const int Mr = r + 8*hi;
      plw[Mr*PSTR + l15]      = f2bf(p0);
      plw[Mr*PSTR + 16 + l15] = f2bf(p1);
    }
    // P as A-frag, preload all 8 V fragments, then 8 WMMAs
    v16bf pf = cat16(ld8(&plw[l15*PSTR + hi*8]), ld8(&plw[l15*PSTR + 16 + hi*8]));
    v16bf vf[8];
    #pragma unroll
    for (int t8 = 0; t8 < 8; ++t8) {
      int base = (t8*16 + l15)*VSTR + hi*8;
      vf[t8] = cat16(ld8(&lv[base]), ld8(&lv[base + 16]));
    }
    #pragma unroll
    for (int t8 = 0; t8 < 8; ++t8)
      acc[t8] = wmma_bf16(pf, vf[t8], acc[t8]);
    __syncthreads();
  }
  #pragma unroll
  for (int t8 = 0; t8 < 8; ++t8) {
    #pragma unroll
    for (int r = 0; r < 8; ++r) {
      const int row = qrow0 + r + 8*hi;
      out[(bs0 + row)*(size_t)H + h*HD + t8*16 + l15] = f2bf(acc[t8][r] / lrow[r]);
    }
  }
}

// ---------------------------------------------------------------- launch
extern "C" void kernel_launch(void* const* d_in, const int* in_sizes, int n_in,
                              void* d_out, int out_size, void* d_ws, size_t ws_size,
                              hipStream_t stream) {
  (void)in_sizes; (void)n_in; (void)out_size; (void)ws_size;
  const long long* pid  = (const long long*)d_in[0];
  const float* hidden   = (const float*)d_in[1];
  const float* ln_w     = (const float*)d_in[2];
  const float* ln_b     = (const float*)d_in[3];
  const float* qkv_w    = (const float*)d_in[4];
  const float* qkv_b    = (const float*)d_in[5];
  const float* dense_w  = (const float*)d_in[6];
  const float* dense_b  = (const float*)d_in[7];
  const float* fc1_w    = (const float*)d_in[8];
  const float* fc1_b    = (const float*)d_in[9];
  const float* fc2_w    = (const float*)d_in[10];
  const float* fc2_b    = (const float*)d_in[11];

  const size_t M = (size_t)B_SZ * S_LEN;   // 4096
  char* ws = (char*)d_ws;
  size_t off = 0;
  auto carve = [&](size_t bytes) -> char* {
    char* p = ws + off;
    off = (off + bytes + 255) & ~(size_t)255;
    return p;
  };
  unsigned short* lnb  = (unsigned short*)carve(M*(size_t)H*2);
  unsigned short* qkvb = (unsigned short*)carve(M*(size_t)H3*2);
  unsigned short* qr   = (unsigned short*)carve(M*(size_t)H*2);
  unsigned short* kr   = (unsigned short*)carve(M*(size_t)H*2);
  unsigned short* attn = (unsigned short*)carve(M*(size_t)H*2);
  float*          dres = (float*)carve(M*(size_t)H*4);
  unsigned short* h1   = (unsigned short*)carve(M*(size_t)FF*2);

  ln_kernel<<<(unsigned)M, 256, 0, stream>>>(hidden, ln_w, ln_b, lnb);

  gemm_bf16_kernel<0><<<dim3(H3/128, (unsigned)(M/128)), 256, 0, stream>>>(
      lnb, qkv_w, qkv_b, nullptr, qkvb, nullptr, nullptr, (int)M, H3, H);

  rope_kernel<<<(unsigned)(M*(size_t)H/256), 256, 0, stream>>>(qkvb, pid, qr, kr);

  attn_kernel<<<dim3(S_LEN/64, NH, B_SZ), 128, 0, stream>>>(qr, kr, qkvb, attn);

  gemm_bf16_kernel<2><<<dim3(H/128, (unsigned)(M/128)), 256, 0, stream>>>(
      attn, dense_w, dense_b, dres, nullptr, nullptr, nullptr, (int)M, H, H);

  gemm_bf16_kernel<1><<<dim3(FF/128, (unsigned)(M/128)), 256, 0, stream>>>(
      lnb, fc1_w, fc1_b, nullptr, h1, nullptr, nullptr, (int)M, FF, H);

  gemm_bf16_kernel<3><<<dim3(H/128, (unsigned)(M/128)), 256, 0, stream>>>(
      h1, fc2_w, fc2_b, (float*)d_out, nullptr, dres, hidden, (int)M, H, FF);
}
